// QuantumInspiredAttention_30442728194645
// MI455X (gfx1250) — matrix operand order repaired and checked by reading, not compiled, not run on hardware
//
#include <hip/hip_runtime.h>
#include <hip/hip_fp16.h>

typedef __attribute__((ext_vector_type(16))) _Float16 v16h;
typedef __attribute__((ext_vector_type(8)))  _Float16 v8h;
typedef __attribute__((ext_vector_type(8)))  float    v8f;

#define DMODEL 4096
#define TOKENS 2048
#define SEQ    1024
#define NHEAD  16
#define MSUP   4
#define DKH    64
#define LDS_K  40   // 32 + 8 halves pad

static __device__ __forceinline__ v8f vzero8() {
    v8f z = {0.f,0.f,0.f,0.f,0.f,0.f,0.f,0.f};
    return z;
}

// async 16-byte global->LDS copy (GVS mode), tracked by ASYNCcnt
static __device__ __forceinline__ void async_ld16(void* lds_dst, const void* gbase, unsigned goff) {
    unsigned l = (unsigned)(size_t)lds_dst;   // low 32 bits of generic addr = LDS offset
    asm volatile("global_load_async_to_lds_b128 %0, %1, %2"
                 :: "v"(l), "v"(goff), "s"(gbase)
                 : "memory");
}

// ---- fragment loaders from LDS (layouts per CDNA5 ISA 7.12.2) ----
static __device__ __forceinline__ v16h load_afrag(const _Float16* As, int m0, int lane) {
    int r  = m0 + (lane & 15);
    int kb = (lane & 16) ? 8 : 0;
    const _Float16* p = As + r * LDS_K + kb;
    v16h a;
    *(v8h*)&a       = *(const v8h*)(p);        // halves 0-7  -> K = kb..kb+7
    *((v8h*)&a + 1) = *(const v8h*)(p + 16);   // halves 8-15 -> K = kb+16..kb+23
    return a;
}

static __device__ __forceinline__ v16h load_bfrag(const _Float16* Ws, int n0, int lane) {
    int n  = n0 + (lane & 15);
    int kb = (lane & 16) ? 8 : 0;
    const _Float16* p = Ws + n * LDS_K + kb;
    v16h b;
    *(v8h*)&b       = *(const v8h*)(p);
    *((v8h*)&b + 1) = *(const v8h*)(p + 16);
    return b;
}

// ---- shared GEMM mainloop: C[128x128] = A[128xK] * W[128xK]^T, reg-prefetch pipelined ----
template <bool AF16>
static __device__ __forceinline__ void gemm_tile(const void* Ag, const float* Wg,
                                                 int m0g, int n0g,
                                                 _Float16* As, _Float16* Ws,
                                                 v8f acc[8], int tid) {
    int lane = tid & 31;
    int wave = tid >> 5;
    int wm = (wave & 1) * 64;
    int wn = (wave >> 1) * 32;
#pragma unroll
    for (int i = 0; i < 8; ++i) acc[i] = vzero8();

    int row = tid >> 1;
    int c   = (tid & 1) * 16;
    const float*    Wp   = Wg + (size_t)(n0g + row) * DMODEL + c;
    const float*    Ap32 = AF16 ? nullptr : ((const float*)Ag + (size_t)(m0g + row) * DMODEL + c);
    const _Float16* Ap16 = AF16 ? ((const _Float16*)Ag + (size_t)(m0g + row) * DMODEL + c) : nullptr;

    float4 aw[4], ww[4];
    v8h    a16[2];

    auto loadRegs = [&](int k0) {
        if (AF16) {
            const v8h* g = (const v8h*)(Ap16 + k0);
            a16[0] = g[0]; a16[1] = g[1];
        } else {
            const float4* g = (const float4*)(Ap32 + k0);
            aw[0] = g[0]; aw[1] = g[1]; aw[2] = g[2]; aw[3] = g[3];
        }
        const float4* w = (const float4*)(Wp + k0);
        ww[0] = w[0]; ww[1] = w[1]; ww[2] = w[2]; ww[3] = w[3];
    };
    auto storeTiles = [&]() {
        _Float16* dA = As + row * LDS_K + c;
        if (AF16) {
            *(v8h*)(dA)     = a16[0];
            *(v8h*)(dA + 8) = a16[1];
        } else {
            v8h h0 = {(_Float16)aw[0].x,(_Float16)aw[0].y,(_Float16)aw[0].z,(_Float16)aw[0].w,
                      (_Float16)aw[1].x,(_Float16)aw[1].y,(_Float16)aw[1].z,(_Float16)aw[1].w};
            v8h h1 = {(_Float16)aw[2].x,(_Float16)aw[2].y,(_Float16)aw[2].z,(_Float16)aw[2].w,
                      (_Float16)aw[3].x,(_Float16)aw[3].y,(_Float16)aw[3].z,(_Float16)aw[3].w};
            *(v8h*)(dA)     = h0;
            *(v8h*)(dA + 8) = h1;
        }
        v8h w0 = {(_Float16)ww[0].x,(_Float16)ww[0].y,(_Float16)ww[0].z,(_Float16)ww[0].w,
                  (_Float16)ww[1].x,(_Float16)ww[1].y,(_Float16)ww[1].z,(_Float16)ww[1].w};
        v8h w1 = {(_Float16)ww[2].x,(_Float16)ww[2].y,(_Float16)ww[2].z,(_Float16)ww[2].w,
                  (_Float16)ww[3].x,(_Float16)ww[3].y,(_Float16)ww[3].z,(_Float16)ww[3].w};
        _Float16* dW = Ws + row * LDS_K + c;
        *(v8h*)(dW)     = w0;
        *(v8h*)(dW + 8) = w1;
    };

    loadRegs(0);
    storeTiles();
    __syncthreads();

    for (int k0 = 0; k0 < DMODEL; k0 += 32) {
        bool nx = (k0 + 32) < DMODEL;
        if (nx) loadRegs(k0 + 32);        // global loads issue; wait lands after WMMAs

        v16h b0 = load_bfrag(Ws, wn,      lane);
        v16h b1 = load_bfrag(Ws, wn + 16, lane);
#pragma unroll
        for (int i = 0; i < 4; ++i) {
            v16h a = load_afrag(As, wm + i * 16, lane);
            acc[i*2+0] = __builtin_amdgcn_wmma_f32_16x16x32_f16(false, a, false, b0,
                            (short)0, acc[i*2+0], false, false);
            acc[i*2+1] = __builtin_amdgcn_wmma_f32_16x16x32_f16(false, a, false, b1,
                            (short)0, acc[i*2+1], false, false);
        }
        __syncthreads();                  // everyone done reading LDS tile
        if (nx) {
            storeTiles();                 // overwrite with prefetched tile
            __syncthreads();
        }
    }
}

// ---- kernel 1: QKV projection + bias + interference mix + scatter to head-major f16 ----
__global__ __launch_bounds__(256) void proj_kernel(
        const float* __restrict__ q, const float* __restrict__ k, const float* __restrict__ v,
        const float* __restrict__ Wq, const float* __restrict__ Wk, const float* __restrict__ Wv,
        const float* __restrict__ bq, const float* __restrict__ bk, const float* __restrict__ bv,
        const float* __restrict__ intf, const float* __restrict__ amp,
        _Float16* __restrict__ qh, _Float16* __restrict__ kh, _Float16* __restrict__ vhT) {
    __shared__ _Float16 As[128 * LDS_K];
    __shared__ _Float16 Ws[128 * LDS_K];
    __shared__ float    Cs[128][132];

    int which = blockIdx.z;
    const float* A    = (which == 0) ? q  : (which == 1) ? k  : v;
    const float* W    = (which == 0) ? Wq : (which == 1) ? Wk : Wv;
    const float* bias = (which == 0) ? bq : (which == 1) ? bk : bv;
    _Float16*    out  = (which == 0) ? qh : (which == 1) ? kh : vhT;
    bool transp = (which == 2);

    int m0g = blockIdx.x * 128;
    int n0g = blockIdx.y * 128;
    int tid = threadIdx.x, lane = tid & 31, wave = tid >> 5;

    v8f acc[8];
    gemm_tile<false>(A, W, m0g, n0g, As, Ws, acc, tid);

    int wm = (wave & 1) * 64, wn = (wave >> 1) * 32;
    int hi8 = (lane & 16) ? 8 : 0;
    __syncthreads();
#pragma unroll
    for (int i = 0; i < 4; ++i)
#pragma unroll
        for (int j = 0; j < 2; ++j)
#pragma unroll
            for (int r = 0; r < 8; ++r) {
                int mm = wm + i * 16 + r + hi8;
                int nn = wn + j * 16 + (lane & 15);
                Cs[mm][nn] = acc[i * 2 + j][r] + bias[n0g + nn];
            }
    __syncthreads();

    // interference: y'_j = y_j + 0.1*amp * sum_i y_i * intf[h][i][j]  (groups of 4 cols)
    int hh = n0g / 256;                   // 128-wide tile stays inside one head
    float a01 = 0.1f * amp[0];
    float W44[16];
#pragma unroll
    for (int i = 0; i < 16; ++i) W44[i] = intf[hh * 16 + i];

    for (int cell = tid; cell < 128 * 32; cell += 256) {
        int row = cell >> 5;
        int g   = cell & 31;
        float y0 = Cs[row][g * 4 + 0];
        float y1 = Cs[row][g * 4 + 1];
        float y2 = Cs[row][g * 4 + 2];
        float y3 = Cs[row][g * 4 + 3];
        float o0 = y0 + a01 * (y0 * W44[0] + y1 * W44[4] + y2 * W44[8]  + y3 * W44[12]);
        float o1 = y1 + a01 * (y0 * W44[1] + y1 * W44[5] + y2 * W44[9]  + y3 * W44[13]);
        float o2 = y2 + a01 * (y0 * W44[2] + y1 * W44[6] + y2 * W44[10] + y3 * W44[14]);
        float o3 = y3 + a01 * (y0 * W44[3] + y1 * W44[7] + y2 * W44[11] + y3 * W44[15]);

        int t = m0g + row;
        int b = t >> 10, s = t & 1023;
        int cglob = n0g + g * 4;                 // = hh*256 + d*4
        int d = (cglob & 255) >> 2;
        size_t hb  = ((size_t)(b * NHEAD + hh) * MSUP) * (SEQ * DKH);  // m=0 block
        size_t off = transp ? ((size_t)d * SEQ + s) : ((size_t)s * DKH + d);
        out[hb + 0 * (SEQ * DKH) + off] = (_Float16)o0;
        out[hb + 1 * (SEQ * DKH) + off] = (_Float16)o1;
        out[hb + 2 * (SEQ * DKH) + off] = (_Float16)o2;
        out[hb + 3 * (SEQ * DKH) + off] = (_Float16)o3;
    }
}

// ---- kernel 2: flash attention per (b,h,m) with async double-buffered K/V staging ----
__global__ __launch_bounds__(128) void attn_kernel(
        const _Float16* __restrict__ qh, const _Float16* __restrict__ kh,
        const _Float16* __restrict__ vhT, _Float16* __restrict__ attnout) {
    __shared__ _Float16 Ks[2][64][72];
    __shared__ _Float16 Vs[2][64][72];
    __shared__ _Float16 Pl[4][16][72];

    int qb = blockIdx.x;            // 16 query blocks of 64
    int hm = blockIdx.y;            // h*4 + m
    int b  = blockIdx.z;
    int hh = hm >> 2, m = hm & 3;
    int tid = threadIdx.x, lane = tid & 31, wave = tid >> 5;
    int hi  = (lane & 16) ? 1 : 0;
    int kb0 = hi ? 8 : 0;

    size_t base = ((size_t)((b * NHEAD + hh) * MSUP + m)) * (SEQ * DKH);
    const _Float16* Q = qh  + base;
    const _Float16* K = kh  + base;          // [s][dk]
    const _Float16* V = vhT + base;          // [dk][s]

    // async prefetch of one 64-key block: K 64x64 f16 + V 64x64 f16, 16B chunks
    auto prefetch = [&](int bsel, int kblk) {
#pragma unroll
        for (int t = 0; t < 4; ++t) {
            int ch  = t * 128 + tid;         // 512 chunks per tile
            int row = ch >> 3;
            int c16 = ch & 7;
            async_ld16(&Ks[bsel][row][c16 * 8], K,
                       (unsigned)((kblk + row) * 128 + c16 * 16));
            async_ld16(&Vs[bsel][row][c16 * 8], V,
                       (unsigned)(row * 2048 + kblk * 2 + c16 * 16));
        }
    };

    int qrow = qb * 64 + wave * 16 + (lane & 15);
    v16h qf[2];
#pragma unroll
    for (int c = 0; c < 2; ++c) {
        const _Float16* p = Q + (size_t)qrow * DKH + c * 32 + kb0;
        *(v8h*)&qf[c]       = *(const v8h*)(p);
        *((v8h*)&qf[c] + 1) = *(const v8h*)(p + 16);
    }

    float mr[8], lr[8];
    v8f o[4];
#pragma unroll
    for (int r = 0; r < 8; ++r) { mr[r] = -1e30f; lr[r] = 0.f; }
#pragma unroll
    for (int j = 0; j < 4; ++j) o[j] = vzero8();

    prefetch(0, 0);
    int buf = 0;
    for (int kblk = 0; kblk < SEQ; kblk += 64) {
        bool has_next = (kblk + 64) < SEQ;
        if (has_next) {
            prefetch(buf ^ 1, kblk + 64);                    // 8 more async ops
            asm volatile("s_wait_asynccnt 0x8" ::: "memory"); // current block's 8 done
        } else {
            asm volatile("s_wait_asynccnt 0x0" ::: "memory");
        }
        __syncthreads();

        v8f st[4];
#pragma unroll
        for (int t = 0; t < 4; ++t) {
            st[t] = vzero8();
#pragma unroll
            for (int c = 0; c < 2; ++c) {
                const _Float16* p = &Ks[buf][t * 16 + (lane & 15)][c * 32 + hi * 8];
                v16h kf;
                *(v8h*)&kf       = *(const v8h*)(p);
                *((v8h*)&kf + 1) = *(const v8h*)(p + 16);
                st[t] = __builtin_amdgcn_wmma_f32_16x16x32_f16(false, qf[c], false, kf,
                            (short)0, st[t], false, false);
            }
        }
        // scale 1/sqrt(dk)
#pragma unroll
        for (int t = 0; t < 4; ++t)
#pragma unroll
            for (int r = 0; r < 8; ++r) st[t][r] *= 0.125f;

        // online softmax per row (rows replicated across each 16-lane half)
#pragma unroll
        for (int r = 0; r < 8; ++r) {
            float tm = fmaxf(fmaxf(st[0][r], st[1][r]), fmaxf(st[2][r], st[3][r]));
#pragma unroll
            for (int off = 8; off >= 1; off >>= 1) tm = fmaxf(tm, __shfl_xor(tm, off, 32));
            float mnew  = fmaxf(mr[r], tm);
            float alpha = __expf(mr[r] - mnew);
            float ps = 0.f;
#pragma unroll
            for (int t = 0; t < 4; ++t) {
                float p = __expf(st[t][r] - mnew);
                st[t][r] = p;
                ps += p;
            }
#pragma unroll
            for (int off = 8; off >= 1; off >>= 1) ps += __shfl_xor(ps, off, 32);
            lr[r] = lr[r] * alpha + ps;
            mr[r] = mnew;
#pragma unroll
            for (int j = 0; j < 4; ++j) o[j][r] *= alpha;
        }

        // P (C-layout) -> LDS -> A-layout fragments (intra-wave, DS ops in-order)
#pragma unroll
        for (int t = 0; t < 4; ++t)
#pragma unroll
            for (int r = 0; r < 8; ++r)
                Pl[wave][r + hi * 8][t * 16 + (lane & 15)] = (_Float16)st[t][r];

        v16h pf[2];
#pragma unroll
        for (int c = 0; c < 2; ++c) {
            const _Float16* p = &Pl[wave][lane & 15][c * 32 + kb0];
            *(v8h*)&pf[c]       = *(const v8h*)(p);
            *((v8h*)&pf[c] + 1) = *(const v8h*)(p + 16);
        }

#pragma unroll
        for (int j = 0; j < 4; ++j) {
            int d = j * 16 + (lane & 15);
#pragma unroll
            for (int c = 0; c < 2; ++c) {
                const _Float16* p = &Vs[buf][d][c * 32 + hi * 8];
                v16h vf;
                *(v8h*)&vf       = *(const v8h*)(p);
                *((v8h*)&vf + 1) = *(const v8h*)(p + 16);
                o[j] = __builtin_amdgcn_wmma_f32_16x16x32_f16(false, pf[c], false, vf,
                            (short)0, o[j], false, false);
            }
        }
        __syncthreads();   // all waves done with buf before it is re-prefetched
        buf ^= 1;
    }

    // epilogue: normalize, scatter to token-major [t][h*256 + d*4 + m] f16
#pragma unroll
    for (int j = 0; j < 4; ++j) {
        int d = j * 16 + (lane & 15);
#pragma unroll
        for (int r = 0; r < 8; ++r) {
            int srow = qb * 64 + wave * 16 + r + hi * 8;
            size_t tok = (size_t)b * SEQ + srow;
            float val = o[j][r] / lr[r];
            attnout[tok * DMODEL + hh * 256 + d * 4 + m] = (_Float16)val;
        }
    }
}

// ---- kernel 3: output projection (f16 A from ws, f32 weights, f32 out) ----
__global__ __launch_bounds__(256) void outproj_kernel(
        const _Float16* __restrict__ attn, const float* __restrict__ Wo,
        const float* __restrict__ bo, float* __restrict__ out) {
    __shared__ _Float16 As[128 * LDS_K];
    __shared__ _Float16 Ws[128 * LDS_K];

    int m0g = blockIdx.x * 128;
    int n0g = blockIdx.y * 128;
    int tid = threadIdx.x, lane = tid & 31, wave = tid >> 5;

    v8f acc[8];
    gemm_tile<true>(attn, Wo, m0g, n0g, As, Ws, acc, tid);

    int wm = (wave & 1) * 64, wn = (wave >> 1) * 32;
    int hi8 = (lane & 16) ? 8 : 0;
#pragma unroll
    for (int i = 0; i < 4; ++i)
#pragma unroll
        for (int j = 0; j < 2; ++j)
#pragma unroll
            for (int r = 0; r < 8; ++r) {
                int mm = wm + i * 16 + r + hi8;
                int nn = wn + j * 16 + (lane & 15);
                out[(size_t)(m0g + mm) * DMODEL + n0g + nn] =
                    acc[i * 2 + j][r] + bo[n0g + nn];
            }
}

extern "C" void kernel_launch(void* const* d_in, const int* in_sizes, int n_in,
                              void* d_out, int out_size, void* d_ws, size_t ws_size,
                              hipStream_t stream) {
    const float* q    = (const float*)d_in[0];
    const float* k    = (const float*)d_in[1];
    const float* v    = (const float*)d_in[2];
    const float* Wq   = (const float*)d_in[3];
    const float* bq   = (const float*)d_in[4];
    const float* Wk   = (const float*)d_in[5];
    const float* bk   = (const float*)d_in[6];
    const float* Wv   = (const float*)d_in[7];
    const float* bv   = (const float*)d_in[8];
    const float* intf = (const float*)d_in[9];
    const float* amp  = (const float*)d_in[10];
    const float* Wo   = (const float*)d_in[11];
    const float* bo   = (const float*)d_in[12];

    size_t n1 = (size_t)TOKENS * DMODEL;   // elements per f16 activation tensor
    _Float16* ws   = (_Float16*)d_ws;
    _Float16* qh   = ws;
    _Float16* kh   = ws + n1;
    _Float16* vhT  = ws + 2 * n1;
    _Float16* attn = ws + 3 * n1;

    dim3 g1(TOKENS / 128, DMODEL / 128, 3);
    proj_kernel<<<g1, 256, 0, stream>>>(q, k, v, Wq, Wk, Wv, bq, bk, bv,
                                        intf, amp, qh, kh, vhT);

    dim3 g2(SEQ / 64, NHEAD * MSUP, 2);
    attn_kernel<<<g2, 128, 0, stream>>>(qh, kh, vhT, attn);

    dim3 g3(TOKENS / 128, DMODEL / 128, 1);
    outproj_kernel<<<g3, 256, 0, stream>>>(attn, Wo, bo, (float*)d_out);
}
